// ClassifierRGCNHidden_72859825209453
// MI455X (gfx1250) — matrix operand order repaired and checked by reading, not compiled, not run on hardware
//
#include <hip/hip_runtime.h>
#include <stdint.h>

// RGCN forward for MI455X (gfx1250): bf16 WMMA GEMMs (register-blocked 32x64
// per wave) for all relation/self-loop transforms, per-relation chunking of the
// [R,N,dout] tensor to bound workspace, float-atomic edge scatter, then
// pooling + tiny MLP head.

#define N_NODES  50000
#define N_EDGES  600000
#define N_RELS   16
#define N_GRAPHS 128

typedef __attribute__((ext_vector_type(16))) __bf16        v16bf;
typedef __attribute__((ext_vector_type(8)))  float         v8f;
typedef __attribute__((ext_vector_type(4)))  unsigned int  u32x4;
typedef __attribute__((ext_vector_type(4)))  float         f32x4;

__device__ __forceinline__ unsigned short f2bf(float f) {
  union { float f; unsigned u; } v; v.f = f;
  unsigned r = v.u + 0x7FFFu + ((v.u >> 16) & 1u); // RNE
  return (unsigned short)(r >> 16);
}

// ---- elementwise: f32 -> bf16 -------------------------------------------------
__global__ void k_conv_bf16(const float* __restrict__ x, unsigned short* __restrict__ y, int n) {
  int i = blockIdx.x * blockDim.x + threadIdx.x;
  if (i < n) y[i] = f2bf(x[i]);
}

// ---- transpose-convert: W [B,K,N] f32 -> WT [B,N,K] bf16 ----------------------
__global__ void k_tconv(const float* __restrict__ W, unsigned short* __restrict__ WT,
                        int B, int K, int Ncols) {
  int i = blockIdx.x * blockDim.x + threadIdx.x;
  int tot = B * K * Ncols;
  if (i >= tot) return;
  int b   = i / (K * Ncols);
  int rem = i - b * (K * Ncols);
  int k   = rem / Ncols;
  int n   = rem - k * Ncols;
  WT[(size_t)b * K * Ncols + (size_t)n * K + k] = f2bf(W[i]);
}

union FragU { u32x4 u4[2]; v16bf v; };

// ---- C[M,N] = A[M,K]bf16 * B (stored transposed: BT[N,K]) (+bias) -------------
// block = 128 threads = 4 waves arranged 2x2; block tile 64(M) x 128(N).
// Each wave owns a 32x64 tile: 2 A-frags x 4 B-frags -> 8 accumulators, so each
// 32-deep K step issues 12 b128 loads for 8 WMMAs (vs 4 loads per WMMA naive).
// A frag (16x32, ISA 7.12.2): lanes 0-15 row m, K = {k..k+7, k+16..k+23};
//                             lanes 16-31 row m, K = {k+8..k+15, k+24..k+31}.
// B frag (32x16): lane n&15, K = {k..k+7, k+16..k+23} (lanes<16) / {k+8..k+15, k+24..k+31} (lanes>=16),
//                 same K order as A; BT is [N,K].
__global__ __launch_bounds__(128) void k_gemm_bf16(
    const unsigned short* __restrict__ A, const unsigned short* __restrict__ BT,
    const float* __restrict__ bias, float* __restrict__ C,
    int M, int K, int Ncols) {
  int lane = threadIdx.x & 31;
  int wave = threadIdx.x >> 5;       // 0..3
  int wrow = wave >> 1;              // 0..1
  int wcol = wave & 1;               // 0..1
  int row0 = blockIdx.y * 64 + wrow * 32;    // wave M-tile base (2 sub-tiles of 16)
  int col0 = blockIdx.x * 128 + wcol * 64;   // wave N-tile base (4 sub-tiles of 16)
  int lm   = lane & 15;
  int kA   = (lane < 16) ? 0 : 8;
  int kB   = (lane < 16) ? 0 : 8;

  // clamp A rows for the (rare) M-edge block; stores are guarded below
  int mA0 = row0 + lm;       if (mA0 > M - 1) mA0 = M - 1;
  int mA1 = row0 + 16 + lm;  if (mA1 > M - 1) mA1 = M - 1;
  const unsigned short* arow0 = A  + (size_t)mA0 * K + kA;
  const unsigned short* arow1 = A  + (size_t)mA1 * K + kA;
  const unsigned short* brow  = BT + (size_t)(col0 + lm) * K + kB;

  v8f acc[2][4] = {};
  for (int k = 0; k < K; k += 32) {
    FragU fa[2], fb[4];
    {
      const u32x4* p0 = reinterpret_cast<const u32x4*>(arow0 + k);
      fa[0].u4[0] = p0[0]; fa[0].u4[1] = p0[2];   // K runs [kA..kA+7], [kA+16..kA+23]
      const u32x4* p1 = reinterpret_cast<const u32x4*>(arow1 + k);
      fa[1].u4[0] = p1[0]; fa[1].u4[1] = p1[2];
    }
#pragma unroll
    for (int j = 0; j < 4; ++j) {
      const u32x4* pb = reinterpret_cast<const u32x4*>(brow + (size_t)j * 16 * K + k);
      fb[j].u4[0] = pb[0]; fb[j].u4[1] = pb[2];   // K runs [kB..kB+7], [kB+16..kB+23]
    }
#pragma unroll
    for (int i = 0; i < 2; ++i)
#pragma unroll
      for (int j = 0; j < 4; ++j)
        acc[i][j] = __builtin_amdgcn_wmma_f32_16x16x32_bf16(
            false, fa[i].v, false, fb[j].v, (short)0, acc[i][j], false, false);
  }

  int mbase = row0 + ((lane < 16) ? 0 : 8);  // C/D layout: VGPR v -> M = v (+8 hi half)
#pragma unroll
  for (int i = 0; i < 2; ++i) {
#pragma unroll
    for (int j = 0; j < 4; ++j) {
      int n = col0 + j * 16 + lm;
      float badd = bias ? bias[n] : 0.0f;
#pragma unroll
      for (int v = 0; v < 8; ++v) {
        int mr = mbase + i * 16 + v;
        if (mr < M) C[(size_t)mr * Ncols + n] = acc[i][j][v] + badd;
      }
    }
  }
}

// ---- edge scatter for one relation: agg[dst] += t_r[src] ----------------------
__global__ void k_scatter_rel(const int* __restrict__ src, const int* __restrict__ dst,
                              const int* __restrict__ rel, int r,
                              const float* __restrict__ t, float* __restrict__ agg,
                              int dout) {
  int chunks = dout >> 2;
  long long gid = (long long)blockIdx.x * blockDim.x + threadIdx.x;
  long long tot = (long long)N_EDGES * chunks;
  if (gid >= tot) return;
  int e  = (int)(gid / chunks);
  int c4 = (int)(gid % chunks) << 2;
  if (rel[e] != r) return;
  f32x4 m = *reinterpret_cast<const f32x4*>(t + (size_t)src[e] * dout + c4);
  float* o = agg + (size_t)dst[e] * dout + c4;
  atomicAdd(o + 0, m[0]);
  atomicAdd(o + 1, m[1]);
  atomicAdd(o + 2, m[2]);
  atomicAdd(o + 3, m[3]);
}

__global__ void k_relu(float* x, int n) {
  int i = blockIdx.x * blockDim.x + threadIdx.x;
  if (i < n) x[i] = fmaxf(x[i], 0.0f);
}

__global__ void k_zero(float* x, int n) {
  int i = blockIdx.x * blockDim.x + threadIdx.x;
  if (i < n) x[i] = 0.0f;
}

// ---- per-graph pooling --------------------------------------------------------
__global__ void k_pool(const float* __restrict__ h, const int* __restrict__ gid,
                       float* __restrict__ pool, float* __restrict__ cnt, int dim) {
  int i = blockIdx.x * blockDim.x + threadIdx.x;
  int tot = N_NODES * dim;
  if (i >= tot) return;
  int node = i / dim;
  int c = i - node * dim;
  int g = gid[node];
  atomicAdd(&pool[(size_t)g * dim + c], h[(size_t)node * dim + c]);
  if (c == 0) atomicAdd(&cnt[g], 1.0f);
}

__global__ void k_pooldiv(const float* __restrict__ pool, const float* __restrict__ cnt,
                          float* __restrict__ hg, int dim) {
  int i = blockIdx.x * blockDim.x + threadIdx.x;
  int tot = N_GRAPHS * dim;
  if (i >= tot) return;
  int g = i / dim;
  hg[i] = pool[i] / fmaxf(cnt[g], 1.0f);
}

// ---- tiny MLP head (~21 MFLOP total) ---------------------------------------
__global__ void k_dense(const float* __restrict__ x, const float* __restrict__ W,
                        const float* __restrict__ b, float* __restrict__ y,
                        int rows, int K, int Ncols, int dorelu) {
  int i = blockIdx.x * blockDim.x + threadIdx.x;
  if (i >= rows * Ncols) return;
  int r = i / Ncols;
  int c = i - r * Ncols;
  float s = b[c];
  for (int k = 0; k < K; ++k) s += x[(size_t)r * K + k] * W[(size_t)k * Ncols + c];
  y[i] = dorelu ? fmaxf(s, 0.0f) : s;
}

__global__ void k_softmax8(const float* __restrict__ logits, float* __restrict__ out) {
  int i = threadIdx.x;
  if (i >= N_GRAPHS) return;
  const float* p = logits + i * 8;
  float mx = p[0];
#pragma unroll
  for (int j = 1; j < 8; ++j) mx = fmaxf(mx, p[j]);
  float e[8], s = 0.0f;
#pragma unroll
  for (int j = 0; j < 8; ++j) { e[j] = __expf(p[j] - mx); s += e[j]; }
#pragma unroll
  for (int j = 0; j < 8; ++j) out[i * 8 + j] = e[j] / s;
}

extern "C" void kernel_launch(void* const* d_in, const int* in_sizes, int n_in,
                              void* d_out, int out_size, void* d_ws, size_t ws_size,
                              hipStream_t stream) {
  (void)in_sizes; (void)n_in; (void)out_size; (void)ws_size;
  const float* h0   = (const float*)d_in[0];
  const int*   src  = (const int*)d_in[1];
  const int*   dst  = (const int*)d_in[2];
  const int*   rel  = (const int*)d_in[3];
  const int*   gids = (const int*)d_in[4];
  const float* Wrel[3]  = {(const float*)d_in[5], (const float*)d_in[8],  (const float*)d_in[11]};
  const float* Wloop[3] = {(const float*)d_in[6], (const float*)d_in[9],  (const float*)d_in[12]};
  const float* bias[3]  = {(const float*)d_in[7], (const float*)d_in[10], (const float*)d_in[13]};
  const float* Wh[3]    = {(const float*)d_in[14], (const float*)d_in[16], (const float*)d_in[18]};
  const float* bh[3]    = {(const float*)d_in[15], (const float*)d_in[17], (const float*)d_in[19]};
  const float* Wc = (const float*)d_in[20];
  const float* bc = (const float*)d_in[21];

  static const int din_[3]  = {128, 128, 256};
  static const int dout_[3] = {128, 256, 128};

  // workspace layout (~182 MB)
  char* ws = (char*)d_ws;
  size_t off = 0;
  auto alloc = [&](size_t bytes) -> char* {
    char* p = ws + off; off += (bytes + 255) & ~(size_t)255; return p;
  };
  float*          bufA   = (float*)alloc((size_t)N_NODES * 256 * 4);
  float*          bufB   = (float*)alloc((size_t)N_NODES * 256 * 4);
  unsigned short* hb     = (unsigned short*)alloc((size_t)N_NODES * 256 * 2);
  float*          t      = (float*)alloc((size_t)N_NODES * 256 * 4);
  unsigned short* wrelT  = (unsigned short*)alloc((size_t)N_RELS * 256 * 256 * 2);
  unsigned short* wloopT = (unsigned short*)alloc((size_t)256 * 256 * 2);
  float*          pool   = (float*)alloc((size_t)N_GRAPHS * 256 * 4);
  float*          cnt    = (float*)alloc((size_t)N_GRAPHS * 4);
  float*          mlpA   = (float*)alloc((size_t)N_GRAPHS * 256 * 4);
  float*          mlpB   = (float*)alloc((size_t)N_GRAPHS * 256 * 4);

  const float* hcur = h0;
  for (int L = 0; L < 3; ++L) {
    int din = din_[L], dout = dout_[L];
    // weights -> transposed bf16 ([dout, din] per matrix, B-fragment friendly)
    int totW = N_RELS * din * dout;
    k_tconv<<<(totW + 255) / 256, 256, 0, stream>>>(Wrel[L], wrelT, N_RELS, din, dout);
    int totL = din * dout;
    k_tconv<<<(totL + 255) / 256, 256, 0, stream>>>(Wloop[L], wloopT, 1, din, dout);
    // activations -> bf16
    int totc = N_NODES * din;
    k_conv_bf16<<<(totc + 255) / 256, 256, 0, stream>>>(hcur, hb, totc);

    float* agg = (hcur == (const float*)bufA) ? bufB : bufA;
    dim3 gg(dout / 128, (N_NODES + 63) / 64);   // block tile 64 x 128

    // agg = h @ W_loop + b   (WMMA)
    k_gemm_bf16<<<gg, 128, 0, stream>>>(hb, wloopT, bias[L], agg, N_NODES, din, dout);

    // per-relation: t_r = h @ W_rel[r] (WMMA), then scatter edges of relation r
    long long etot = (long long)N_EDGES * (dout / 4);
    int eblocks = (int)((etot + 255) / 256);
    for (int r = 0; r < N_RELS; ++r) {
      k_gemm_bf16<<<gg, 128, 0, stream>>>(hb, wrelT + (size_t)r * din * dout,
                                          nullptr, t, N_NODES, din, dout);
      k_scatter_rel<<<eblocks, 256, 0, stream>>>(src, dst, rel, r, t, agg, dout);
    }
    int n = N_NODES * dout;
    k_relu<<<(n + 255) / 256, 256, 0, stream>>>(agg, n);
    hcur = agg;
  }

  // mean pooling over nodes per graph (final dim = 128)
  int dim = 128;
  k_zero<<<(N_GRAPHS * dim + 255) / 256, 256, 0, stream>>>(pool, N_GRAPHS * dim);
  k_zero<<<1, 256, 0, stream>>>(cnt, N_GRAPHS);
  int ptot = N_NODES * dim;
  k_pool<<<(ptot + 255) / 256, 256, 0, stream>>>(hcur, gids, pool, cnt, dim);
  k_pooldiv<<<(N_GRAPHS * dim + 255) / 256, 256, 0, stream>>>(pool, cnt, mlpA, dim);

  // MLP head + classifier + softmax
  int hk[3] = {128, 128, 256};
  int hn[3] = {128, 256, 128};
  float* x = mlpA; float* y = mlpB;
  for (int i = 0; i < 3; ++i) {
    int tot = N_GRAPHS * hn[i];
    k_dense<<<(tot + 255) / 256, 256, 0, stream>>>(x, Wh[i], bh[i], y,
                                                   N_GRAPHS, hk[i], hn[i], 1);
    float* tm = x; x = y; y = tm;
  }
  k_dense<<<(N_GRAPHS * 8 + 255) / 256, 256, 0, stream>>>(x, Wc, bc, y,
                                                          N_GRAPHS, 128, 8, 0);
  k_softmax8<<<1, 128, 0, stream>>>(y, (float*)d_out);
}
